// VectorQuantizerEMA_21320217657914
// MI455X (gfx1250) — matrix operand order repaired and checked by reading, not compiled, not run on hardware
//
#include <hip/hip_runtime.h>

typedef __attribute__((ext_vector_type(16))) _Float16 v16h;
typedef __attribute__((ext_vector_type(8)))  float    v8f;

#define D_DIM   64
#define K_CODES 512
#define N_ROWS  131072
#define HW      4096
#define CHW     (64*4096)

// workspace layout (bytes)
#define WS_FRAG 0               // 32768 f16 = 65536 B (codebook in B-fragment order)
#define WS_E2   65536           // 512 f32 = 2048 B    (||e_k||^2)
#define WS_BSUM 67584           // 1024 f32 (per-block squared-error partials)

// dynamic LDS layout (bytes)
#define SM_FRAG 0               // 65536
#define SM_E2   65536           // 2048
#define SM_IDX  67584           // 8 waves * 16 ints = 512
#define SM_RED  68096           // 256 f32 = 1024
#define SM_SIZE 69120

// ---------------------------------------------------------------------------
// Async global->LDS copy (gfx1250): 16B per lane, tracked by ASYNCcnt.
// ---------------------------------------------------------------------------
__device__ __forceinline__ void async_copy_b128(unsigned lds_addr, const void* gsrc) {
    asm volatile("global_load_async_to_lds_b128 %0, %1, off"
                 :: "v"(lds_addr), "v"((unsigned long long)(uintptr_t)gsrc)
                 : "memory");
}

// ---------------------------------------------------------------------------
// Kernel 1: repack codebook into WMMA B-fragment layout (f16) + ||e||^2.
// B for V_WMMA_F32_16X16X32_F16 (wave32): 32x16 (KxN); lane<16: e<8 -> K=e,
// e>=8 -> K=e+8; lane>=16: +8; N = lane&15.
// halfIdx = ((chunk*32 + tile)*32 + lane)*16 + e.
// ---------------------------------------------------------------------------
__global__ void vq_prep(const float* __restrict__ cb, unsigned char* __restrict__ ws) {
    _Float16* frag = (_Float16*)(ws + WS_FRAG);
    float*    e2   = (float*)(ws + WS_E2);
    int tid = blockIdx.x * blockDim.x + threadIdx.x;      // 32768 threads
    int c   = tid >> 14;
    int rem = tid & 16383;
    int t   = rem >> 9;
    int l   = (rem >> 4) & 31;
    int e   = rem & 15;
    int kb   = ((e < 8) ? e : e + 8) + ((l < 16) ? 0 : 8);
    int code = t * 16 + (l & 15);
    frag[tid] = (_Float16)cb[code * D_DIM + c * 32 + kb];
    if (tid < K_CODES) {
        float s = 0.f;
        #pragma unroll 8
        for (int d = 0; d < D_DIM; ++d) { float v = cb[tid * D_DIM + d]; s += v * v; }
        e2[tid] = s;
    }
}

// ---------------------------------------------------------------------------
// Kernel 2 (fused): WMMA argmin + codebook gather + NCHW output + loss partials.
// Block = 256 threads (8 waves); each wave owns 16 rows -> 128 rows/block.
// x is read exactly once (kept in f32 registers through the argmin), out is
// written once -> traffic ~= HBM floor.
// ---------------------------------------------------------------------------
__global__ void vq_fused(const float* __restrict__ x, const float* __restrict__ cb,
                         const unsigned char* __restrict__ ws,
                         float* __restrict__ out_q, float* __restrict__ out_idx_f,
                         float* __restrict__ bsum) {
    extern __shared__ char smem[];
    float* se2  = (float*)(smem + SM_E2);
    int*   sidx = (int*)(smem + SM_IDX);
    float* red  = (float*)(smem + SM_RED);

    // ---- async stage: codebook fragments (64KB) + ||e||^2 (2KB) into LDS ----
    {
        const unsigned smb = (unsigned)(uintptr_t)smem;
        #pragma unroll
        for (int j = 0; j < 16; ++j) {
            const int off = (j * 256 + (int)threadIdx.x) * 16;
            async_copy_b128(smb + SM_FRAG + off, ws + WS_FRAG + off);
        }
        if (threadIdx.x < 128) {
            const int off = (int)threadIdx.x * 16;
            async_copy_b128(smb + SM_E2 + off, ws + WS_E2 + off);
        }
        asm volatile("s_wait_asynccnt 0" ::: "memory");
    }
    __syncthreads();

    const int  lane = threadIdx.x & 31;
    const int  wave = threadIdx.x >> 5;
    const bool hiH  = lane >= 16;
    const int  M    = lane & 15;
    const int  r0   = blockIdx.x * 128 + wave * 16;    // first row of this wave
    const int  n    = r0 >> 12;                        // image (128 | 4096 spatial)
    const int  row  = r0 + M;                          // A-matrix row
    const int  s    = row & (HW - 1);
    const float* xb = x + (size_t)n * CHW + s;         // dim k -> xb[k*HW]

    // ---- load x once: keep f32 for output/loss, f16 fragments for WMMA ----
    // A layout: lane<16: e<8 -> K=e, e>=8 -> K=e+8; lane>=16: +8.
    float xf0[16], xf1[16];
    v16h a0, a1;
    #pragma unroll
    for (int e = 0; e < 16; ++e) {
        const int K = ((e < 8) ? e : e + 8) + (hiH ? 8 : 0);
        xf0[e] = xb[(size_t)K * HW];
        xf1[e] = xb[(size_t)(K + 32) * HW];
        a0[e] = (_Float16)xf0[e];
        a1[e] = (_Float16)xf1[e];
    }

    float minv[8];
    int   mini[8];
    #pragma unroll
    for (int r = 0; r < 8; ++r) { minv[r] = 3.4e38f; mini[r] = 0; }

    // ---- 32 code tiles x 2 K-chunks of v_wmma_f32_16x16x32_f16 ----
    const int nclass = lane & 15;                      // C/D col N
    #pragma unroll 4
    for (int t = 0; t < 32; ++t) {
        const v16h b0 = *(const v16h*)(smem + (size_t)((0 * 32 + t) * 32 + lane) * 32);
        const v16h b1 = *(const v16h*)(smem + (size_t)((1 * 32 + t) * 32 + lane) * 32);
        v8f acc = {};
        acc = __builtin_amdgcn_wmma_f32_16x16x32_f16(false, a0, false, b0,
                                                     (short)0, acc, false, false);
        acc = __builtin_amdgcn_wmma_f32_16x16x32_f16(false, a1, false, b1,
                                                     (short)0, acc, false, false);
        const int   code = t * 16 + nclass;
        const float e2v  = se2[code];
        #pragma unroll
        for (int r = 0; r < 8; ++r) {
            float dd = e2v - 2.0f * acc[r];            // ||e||^2 - 2 x.e
            if (dd < minv[r]) { minv[r] = dd; mini[r] = code; }
        }
    }

    // ---- min-reduce across the 16 lanes of each half (rows r / 8+r) ----
    #pragma unroll
    for (int m = 1; m < 16; m <<= 1) {
        #pragma unroll
        for (int r = 0; r < 8; ++r) {
            float ov = __shfl_xor(minv[r], m, 32);
            int   oi = __shfl_xor(mini[r], m, 32);
            if (ov < minv[r] || (ov == minv[r] && oi < mini[r])) { minv[r] = ov; mini[r] = oi; }
        }
    }

    // ---- exchange winners through LDS so every lane sees its row's index ----
    if (nclass == 0) {
        const int base = wave * 16 + (hiH ? 8 : 0);
        #pragma unroll
        for (int r = 0; r < 8; ++r) sidx[base + r] = mini[r];
    }
    __syncthreads();

    const int k = sidx[wave * 16 + M];
    if (!hiH) out_idx_f[row] = (float)k;

    // ---- gather codebook row (f32, L2-resident), write NCHW out, accumulate loss ----
    const float* cbr = cb + (size_t)k * D_DIM;
    float* ob = out_q + (size_t)n * CHW + s;
    float sum = 0.f;
    #pragma unroll
    for (int e = 0; e < 16; ++e) {
        const int K = ((e < 8) ? e : e + 8) + (hiH ? 8 : 0);
        const float q0 = cbr[K];
        const float q1 = cbr[K + 32];
        ob[(size_t)K * HW]        = q0;
        ob[(size_t)(K + 32) * HW] = q1;
        const float d0 = q0 - xf0[e];
        const float d1 = q1 - xf1[e];
        sum += d0 * d0 + d1 * d1;
    }

    // ---- block reduction of squared error ----
    red[threadIdx.x] = sum;
    __syncthreads();
    for (int off = 128; off > 0; off >>= 1) {
        if (threadIdx.x < (unsigned)off) red[threadIdx.x] += red[threadIdx.x + off];
        __syncthreads();
    }
    if (threadIdx.x == 0) bsum[blockIdx.x] = red[0];
}

// ---------------------------------------------------------------------------
// Kernel 3: loss = (1 + 0.25) * mean((q - x)^2) over 1024 block partials.
// ---------------------------------------------------------------------------
__global__ void vq_loss(const float* __restrict__ bsum, float* __restrict__ loss) {
    __shared__ float red[256];
    red[threadIdx.x] = bsum[threadIdx.x] + bsum[threadIdx.x + 256]
                     + bsum[threadIdx.x + 512] + bsum[threadIdx.x + 768];
    __syncthreads();
    for (int off = 128; off > 0; off >>= 1) {
        if (threadIdx.x < (unsigned)off) red[threadIdx.x] += red[threadIdx.x + off];
        __syncthreads();
    }
    if (threadIdx.x == 0) loss[0] = red[0] * (1.25f / 8388608.0f);
}

extern "C" void kernel_launch(void* const* d_in, const int* in_sizes, int n_in,
                              void* d_out, int out_size, void* d_ws, size_t ws_size,
                              hipStream_t stream) {
    const float* x  = (const float*)d_in[0];   // (32,64,64,64) f32 NCHW
    const float* cb = (const float*)d_in[1];   // (512,64) f32
    float* outp     = (float*)d_out;
    float* loss     = outp;                           // [1]
    float* out_q    = outp + 1;                       // [8388608] NCHW
    float* out_idx  = outp + 1 + (size_t)CHW * 32;    // [131072]
    unsigned char* ws = (unsigned char*)d_ws;

    vq_prep <<<128,  256, 0,       stream>>>(cb, ws);
    vq_fused<<<1024, 256, SM_SIZE, stream>>>(x, cb, ws, out_q, out_idx,
                                             (float*)(ws + WS_BSUM));
    vq_loss <<<1,    256, 0,       stream>>>((const float*)(ws + WS_BSUM), loss);
}
